// Propagation_85315230368231
// MI455X (gfx1250) — matrix operand order repaired and checked by reading, not compiled, not run on hardware
//
#include <hip/hip_runtime.h>
#include <hip/hip_bf16.h>

typedef __attribute__((ext_vector_type(16))) _Float16 v16h;
typedef __attribute__((ext_vector_type(8)))  _Float16 v8h;
typedef __attribute__((ext_vector_type(8)))  float    v8f;

#define N_PIX   102400          // B*H*W
#define NSEED   4
#define M_TOTAL (N_PIX * NSEED) // 409600 rows
#define TM      64              // rows per tile
#define NTILES  (M_TOTAL / TM)  // 6400
#define EDIM    128

// padded K strides (in halfs) — keep rows 16B aligned and bank-conflict-free
#define SK1 104  // GEMM1 K (72 -> pad 96, stride 104)
#define SK2 136  // GEMM2 K (128, stride 136)
#define SK3 168  // GEMM3 K (159 -> pad 160, stride 168)

// ---- LDS layout (bytes) ----
#define OFF_W1H 0
#define SZ_W1   (128 * SK1 * 2)
#define OFF_W1L (OFF_W1H + SZ_W1)
#define OFF_W2H (OFF_W1L + SZ_W1)
#define SZ_W2   (128 * SK2 * 2)
#define OFF_W2L (OFF_W2H + SZ_W2)
#define OFF_WPH (OFF_W2L + SZ_W2)
#define SZ_WP   (128 * SK3 * 2)
#define OFF_WPL (OFF_WPH + SZ_WP)
// activation A-plane (hi & lo): gather [64][SK1] aliased with GEMM3 input [64][SK3]
#define SZ_APL   (TM * SK3 * 2)
#define OFF_AH   (OFF_WPL + SZ_WP)
#define OFF_AL   (OFF_AH + SZ_APL)
// hidden-plane (hi & lo): [64][SK2]
#define SZ_HPL   (TM * SK2 * 2)
#define OFF_HH   (OFF_AL + SZ_APL)
#define OFF_HL   (OFF_HH + SZ_HPL)
#define LDS_BYTES (OFF_HL + SZ_HPL)   // = 286720 <= 327680
static_assert(LDS_BYTES <= 320 * 1024, "LDS budget");

__device__ __forceinline__ v8f wmma_step(v16h a, v16h b, v8f c) {
    // 8 args: (neg_a, A, neg_b, B, c_mod, C, reuse_a, reuse_b)
    return __builtin_amdgcn_wmma_f32_16x16x32_f16(false, a, false, b, (short)0, c, false, false);
}

// A fragment (ISA 16x32 layout): per-lane halves at [kbase..kbase+7] and [kbase+16..kbase+23]
__device__ __forceinline__ v16h a_frag(const _Float16* __restrict__ p) {
    v8h a0 = *(const v8h*)p;         // 16B aligned
    v8h a1 = *(const v8h*)(p + 16);  // 16B aligned
    return __builtin_shufflevector(a0, a1, 0, 1, 2, 3, 4, 5, 6, 7,
                                           8, 9, 10, 11, 12, 13, 14, 15);
}

// B fragment (ISA 16x32 layout): per-lane halves at [kbase..kbase+7] and [kbase+16..kbase+23] (row of transposed weight)
__device__ __forceinline__ v16h b_frag(const _Float16* __restrict__ p) {
    v8h b0 = *(const v8h*)p;
    v8h b1 = *(const v8h*)(p + 16);
    return __builtin_shufflevector(b0, b1, 0, 1, 2, 3, 4, 5, 6, 7,
                                           8, 9, 10, 11, 12, 13, 14, 15);
}

// split-once fp32 -> (hi, lo) f16 store
__device__ __forceinline__ void store_hilo(_Float16* __restrict__ ph,
                                           _Float16* __restrict__ pl,
                                           int idx, float x) {
    _Float16 h = (_Float16)x;
    ph[idx] = h;
    pl[idx] = (_Float16)(x - (float)h);
}

// One 64x16 output stripe: acc[4] over KSTEPS k-steps, 3 WMMAs/step (hi*hi + lo*hi + hi*lo)
template <int KSTEPS, int SA, int SB>
__device__ __forceinline__ void gemm_stripe(const _Float16* __restrict__ Ah,
                                            const _Float16* __restrict__ Al,
                                            const _Float16* __restrict__ Bh,
                                            const _Float16* __restrict__ Bl,
                                            int lm, int ksel, int col, v8f acc[4]) {
    for (int ks = 0; ks < KSTEPS; ++ks) {
        const int bofs = col * SB + ks * 32 + ksel * 8;
        v16h bh = b_frag(Bh + bofs);
        v16h bl = b_frag(Bl + bofs);
        const int kbase = ks * 32 + ksel * 8;
#pragma unroll
        for (int rb = 0; rb < 4; ++rb) {
            const int aofs = (rb * 16 + lm) * SA + kbase;
            v16h ah = a_frag(Ah + aofs);
            v16h al = a_frag(Al + aofs);
            acc[rb] = wmma_step(ah, bh, acc[rb]);
            acc[rb] = wmma_step(al, bh, acc[rb]);
            acc[rb] = wmma_step(ah, bl, acc[rb]);
        }
    }
}

__device__ __forceinline__ float gelu_exact(float x) {
    return 0.5f * x * (1.0f + erff(x * 0.70710678118654752f));
}

__global__ void __launch_bounds__(256, 1)
propagation_kernel(const float* __restrict__ cv,    // [N_PIX, 8, 64]
                   const int*   __restrict__ seeds, // [N_PIX, 4]
                   const float* __restrict__ W1,    // [72, 128]
                   const float* __restrict__ b1,    // [128]
                   const float* __restrict__ W2,    // [128, 128]
                   const float* __restrict__ b2,    // [128]
                   const float* __restrict__ Wp,    // [159, 128]
                   float* __restrict__ outE,        // [M_TOTAL, 128]
                   float* __restrict__ outL)        // [M_TOTAL]
{
    extern __shared__ char smem[];
    _Float16* sW1h = (_Float16*)(smem + OFF_W1H);
    _Float16* sW1l = (_Float16*)(smem + OFF_W1L);
    _Float16* sW2h = (_Float16*)(smem + OFF_W2H);
    _Float16* sW2l = (_Float16*)(smem + OFF_W2L);
    _Float16* sWph = (_Float16*)(smem + OFF_WPH);
    _Float16* sWpl = (_Float16*)(smem + OFF_WPL);
    _Float16* aH   = (_Float16*)(smem + OFF_AH);   // cost [r][SK1] then GEMM3 input [r][SK3]
    _Float16* aL   = (_Float16*)(smem + OFF_AL);
    _Float16* hH   = (_Float16*)(smem + OFF_HH);   // hidden [r][SK2]
    _Float16* hL   = (_Float16*)(smem + OFF_HL);

    const int t    = threadIdx.x;
    const int lane = t & 31;
    const int wave = t >> 5;      // 0..7 -> 16-column stripe
    const int lm   = lane & 15;
    const int ksel = lane >> 4;   // 0/1 selects K-half per ISA fragment layout
    const int col  = wave * 16 + lm;

    // ---- One-time: transpose + hi/lo-split weights into LDS ----
    for (int idx = t; idx < 128 * 96; idx += 256) {
        int n = idx / 96, k = idx - n * 96;
        float w = (k < 72) ? W1[k * 128 + n] : 0.0f;
        store_hilo(sW1h, sW1l, n * SK1 + k, w);
    }
    for (int idx = t; idx < 128 * 128; idx += 256) {
        int n = idx >> 7, k = idx & 127;
        store_hilo(sW2h, sW2l, n * SK2 + k, W2[k * 128 + n]);
    }
    for (int idx = t; idx < 128 * 160; idx += 256) {
        int n = idx / 160, k = idx - n * 160;
        float w = (k < 159) ? Wp[k * 128 + n] : 0.0f;
        store_hilo(sWph, sWpl, n * SK3 + k, w);
    }
    const float bias1 = b1[col];
    const float bias2 = b2[col];
    __syncthreads();

    const int r    = t >> 2;  // 0..63: row within tile
    const int part = t & 3;
    const float NORM = (float)(3.14 / 64.0);

    for (int tile = blockIdx.x; tile < NTILES; tile += gridDim.x) {
        const int rowg  = tile * TM + r;
        const int n_pix = rowg >> 2;           // rowg = n_pix*4 + s
        const int seed  = seeds[rowg];

        // ---- Stage 0: gather cost [64][72] (k = g*9 + j) hi/lo, pad to 96 ----
#pragma unroll
        for (int gg = 0; gg < 2; ++gg) {
            const int g = part + gg * 4;
            const float* src = cv + ((size_t)n_pix * 8 + g) * 64;
            const int dofs = r * SK1 + g * 9;
#pragma unroll
            for (int j = 0; j < 9; ++j) {
                int d = seed - 4 + j;
                d = d < 0 ? 0 : (d > 63 ? 63 : d);
                store_hilo(aH, aL, dofs + j, src[d]);
            }
        }
#pragma unroll
        for (int u = 0; u < 6; ++u) {
            const int idx = r * SK1 + 72 + part * 6 + u;
            aH[idx] = (_Float16)0.0f;
            aL[idx] = (_Float16)0.0f;
        }
        if (part == 0) outL[rowg] = (float)seed;
        if (part == 1 && tile + (int)gridDim.x < NTILES)
            __builtin_prefetch(seeds + (tile + gridDim.x) * TM + r, 0, 0);
        __syncthreads();

        // ---- Stage 1: GEMM1 [64,96]x[96,128] + bias + GELU -> hidden hi/lo ----
        {
            v8f acc[4] = {};
            gemm_stripe<3, SK1, SK1>(aH, aL, sW1h, sW1l, lm, ksel, col, acc);
#pragma unroll
            for (int rb = 0; rb < 4; ++rb)
#pragma unroll
                for (int i = 0; i < 8; ++i)
                    store_hilo(hH, hL, (rb * 16 + ksel * 8 + i) * SK2 + col,
                               gelu_exact(acc[rb][i] + bias1));
        }
        __syncthreads();

        // ---- Stage 2a: Fourier embedding -> A-plane cols 128..159 ----
        {
            const float coord = (float)seed * NORM;
#pragma unroll
            for (int u = 0; u < 8; ++u) {
                const int c = 128 + part * 8 + u;
                const int j = c - 128;
                float v;
                if (j < 15)       v = __builtin_sinf(coord * (float)(1 << j));
                else if (j < 30)  v = __builtin_cosf(coord * (float)(1 << (j - 15)));
                else if (j == 30) v = coord;
                else              v = 0.0f;   // K pad at 159
                store_hilo(aH, aL, r * SK3 + c, v);
            }
        }
        // ---- Stage 2b: GEMM2 [64,128]x[128,128] + bias -> A-plane cols 0..127 ----
        {
            v8f acc[4] = {};
            gemm_stripe<4, SK2, SK2>(hH, hL, sW2h, sW2l, lm, ksel, col, acc);
#pragma unroll
            for (int rb = 0; rb < 4; ++rb)
#pragma unroll
                for (int i = 0; i < 8; ++i)
                    store_hilo(aH, aL, (rb * 16 + ksel * 8 + i) * SK3 + col,
                               acc[rb][i] + bias2);
        }
        __syncthreads();

        // ---- Stage 3: GEMM3 [64,160]x[160,128] -> global ----
        {
            v8f acc[4] = {};
            gemm_stripe<5, SK3, SK3>(aH, aL, sWph, sWpl, lm, ksel, col, acc);
            float* op = outE + (size_t)tile * TM * EDIM;
#pragma unroll
            for (int rb = 0; rb < 4; ++rb)
#pragma unroll
                for (int i = 0; i < 8; ++i)
                    op[(size_t)(rb * 16 + ksel * 8 + i) * EDIM + col] = acc[rb][i];
        }
        __syncthreads();
    }
}

extern "C" void kernel_launch(void* const* d_in, const int* in_sizes, int n_in,
                              void* d_out, int out_size, void* d_ws, size_t ws_size,
                              hipStream_t stream) {
    (void)in_sizes; (void)n_in; (void)out_size; (void)d_ws; (void)ws_size;
    const float* cv    = (const float*)d_in[0];
    const int*   seeds = (const int*)  d_in[1];
    // d_in[2] = context (unused by reference)
    const float* W1 = (const float*)d_in[3];
    const float* b1 = (const float*)d_in[4];
    const float* W2 = (const float*)d_in[5];
    const float* b2 = (const float*)d_in[6];
    const float* Wp = (const float*)d_in[7];

    float* outE = (float*)d_out;
    float* outL = outE + (size_t)M_TOTAL * EDIM;

    (void)hipFuncSetAttribute((const void*)propagation_kernel,
                              hipFuncAttributeMaxDynamicSharedMemorySize, LDS_BYTES);
    propagation_kernel<<<dim3(320), dim3(256), LDS_BYTES, stream>>>(
        cv, seeds, W1, b1, W2, b2, Wp, outE, outL);
}
